// TwoStageDecoder_14379550507113
// MI455X (gfx1250) — compile-verified, matrix operand order checked by reading, not run on hardware
//
#include <hip/hip_runtime.h>
#include <math.h>

// ---------------- problem constants (from reference) ----------------
#define QN    10000      // 100*100 BEV queries
#define HBD   100
#define WBD   100
#define DCH   128
#define PPT   4          // P
#define LLV   4          // L
#define NCAM  6
#define KCONV 1152       // 9*128 implicit-im2col K
#define RXY   0.250001f  // 0.25*scale + 1e-6
#define RZ    4.000001f  // 4.0 + 1e-6

typedef __attribute__((ext_vector_type(16))) __bf16 v16bf;
typedef __attribute__((ext_vector_type(8)))  float  v8f;

union Frag { unsigned int u[8]; v16bf v; };

__device__ __forceinline__ unsigned short f2bf(float f) {
  unsigned int u = __float_as_uint(f);
  unsigned int r = u + 0x7FFFu + ((u >> 16) & 1u);   // round-to-nearest-even
  return (unsigned short)(r >> 16);
}
__device__ __forceinline__ float bf2f(unsigned short h) {
  return __uint_as_float(((unsigned int)h) << 16);
}

// ---------------- weight / layout conversion kernels ----------------
__global__ void f32_to_bf16_kernel(const float* __restrict__ in,
                                   unsigned short* __restrict__ out, int n) {
  int i = blockIdx.x * blockDim.x + threadIdx.x;
  if (i < n) out[i] = f2bf(in[i]);
}

// in_w (O,I,3,3) -> Wc[o][tap*128 + i]  (tap-major, channel-minor K order)
__global__ void reorder_convw_kernel(const float* __restrict__ in,
                                     unsigned short* __restrict__ out) {
  int idx = blockIdx.x * blockDim.x + threadIdx.x;
  if (idx >= DCH * KCONV) return;
  int o = idx / KCONV;
  int rem = idx % KCONV;
  int t = rem >> 7;          // tap 0..8
  int c = rem & 127;         // in channel
  out[idx] = f2bf(in[(size_t)(o * DCH + c) * 9 + t]);
}

// bev_query (C-major [c][q]) -> pixel-major bf16 image [q][c]
__global__ void img_to_bf16_kernel(const float* __restrict__ in,
                                   unsigned short* __restrict__ out) {
  int idx = blockIdx.x * blockDim.x + threadIdx.x;
  if (idx >= DCH * QN) return;
  int c = idx / QN, q = idx % QN;
  out[(size_t)q * DCH + c] = f2bf(in[idx]);
}

// ---------------- bf16 WMMA GEMM:  out[M][N] = W[M][K] * act[N][K]^T + bias ----------------
// Wave tile = 64(M) x 16(N): B (activation) fragment loaded ONCE per K-step and
// reused across 4 A fragments -> 4 independent WMMAs (no D->A/B hazard chain).
// act_mode: 0 none, 1 relu, 2 exact gelu. Optional pixel-major f32 add.
// outBF: pixel-major [n][M] bf16 (feeds next GEMM); outF32: channel-major [m][N].
__global__ void gemm_bf16_kernel(const unsigned short* __restrict__ W,
                                 const unsigned short* __restrict__ act,
                                 const float* __restrict__ bias,
                                 const float* __restrict__ addF32,
                                 unsigned short* __restrict__ outBF,
                                 float* __restrict__ outF32,
                                 int M, int N, int K, int act_mode) {
  int gw   = (int)((blockIdx.x * blockDim.x + threadIdx.x) >> 5);
  int lane = threadIdx.x & 31;
  int ntiles = N >> 4;
  int total  = (M >> 6) * ntiles;          // M tiles of 64
  if (gw >= total) return;                 // wave-uniform: EXEC all-ones at WMMA
  int m0 = (gw / ntiles) << 6;
  int n0 = (gw % ntiles) << 4;
  int lo = lane & 15, hi = lane >> 4;
  int n = n0 + lo;

  v8f acc[4];
#pragma unroll
  for (int t = 0; t < 4; ++t)
#pragma unroll
    for (int r = 0; r < 8; ++r) acc[t][r] = bias[m0 + t * 16 + 8 * hi + r];

  const unsigned short* wbase = W   + (size_t)(m0 + lo) * K + 8 * hi;
  const unsigned short* arow  = act + (size_t)n * K + 8 * hi;
  for (int k0 = 0; k0 < K; k0 += 32) {
    Frag B;
    *(uint4*)&B.u[0] = *(const uint4*)(arow + k0);
    *(uint4*)&B.u[4] = *(const uint4*)(arow + k0 + 16);
#pragma unroll
    for (int t = 0; t < 4; ++t) {
      Frag A;
      const unsigned short* wr = wbase + (size_t)t * 16 * K + k0;
      *(uint4*)&A.u[0] = *(const uint4*)(wr);
      *(uint4*)&A.u[4] = *(const uint4*)(wr + 16);
      acc[t] = __builtin_amdgcn_wmma_f32_16x16x32_bf16(false, A.v, false, B.v,
                                                       (short)0, acc[t], false, false);
    }
  }

#pragma unroll
  for (int t = 0; t < 4; ++t) {
    int mbase = m0 + t * 16 + 8 * hi;
    if (addF32) {
#pragma unroll
      for (int r = 0; r < 8; ++r) acc[t][r] += addF32[(size_t)n * M + mbase + r];
    }
    if (act_mode == 1) {
#pragma unroll
      for (int r = 0; r < 8; ++r) acc[t][r] = fmaxf(acc[t][r], 0.f);
    } else if (act_mode == 2) {
#pragma unroll
      for (int r = 0; r < 8; ++r)
        acc[t][r] = 0.5f * acc[t][r] * (1.f + erff(acc[t][r] * 0.70710678118654752f));
    }
    if (outF32) {
#pragma unroll
      for (int r = 0; r < 8; ++r) outF32[(size_t)(mbase + r) * N + n] = acc[t][r];
    }
    if (outBF) {
      unsigned int pk0 = (unsigned int)f2bf(acc[t][0]) | ((unsigned int)f2bf(acc[t][1]) << 16);
      unsigned int pk1 = (unsigned int)f2bf(acc[t][2]) | ((unsigned int)f2bf(acc[t][3]) << 16);
      unsigned int pk2 = (unsigned int)f2bf(acc[t][4]) | ((unsigned int)f2bf(acc[t][5]) << 16);
      unsigned int pk3 = (unsigned int)f2bf(acc[t][6]) | ((unsigned int)f2bf(acc[t][7]) << 16);
      uint4 s; s.x = pk0; s.y = pk1; s.z = pk2; s.w = pk3;
      *(uint4*)(outBF + (size_t)n * M + mbase) = s;
    }
  }
}

// ---------------- implicit-im2col 3x3 conv as WMMA GEMM (K=1152) ----------------
// Wave tile = 64(M) x 16(N): the expensive bounds-checked im2col B fragment is
// built once per K-step and reused across 4 weight fragments.
__device__ __forceinline__ void conv_chunk(const unsigned short* __restrict__ img,
                                           int y, int x, int kb, unsigned int* dst) {
  int tap = kb >> 7;                 // 8 consecutive k stay within one tap
  int cc  = kb & 127;
  int py = y + tap / 3 - 1;
  int px = x + tap % 3 - 1;
  if (py >= 0 && py < HBD && px >= 0 && px < WBD) {
    *(uint4*)dst = *(const uint4*)(img + ((size_t)py * WBD + px) * DCH + cc);
  } else {
    dst[0] = dst[1] = dst[2] = dst[3] = 0u;   // zero padding
  }
}

__global__ void gemm_conv3x3_kernel(const unsigned short* __restrict__ Wc,
                                    const unsigned short* __restrict__ img,
                                    const float* __restrict__ bias,
                                    float* __restrict__ outF) {
  int gw   = (int)((blockIdx.x * blockDim.x + threadIdx.x) >> 5);
  int lane = threadIdx.x & 31;
  const int ntiles = QN >> 4;               // 625
  if (gw >= (DCH >> 6) * ntiles) return;    // 2 * 625 waves
  int m0 = (gw / ntiles) << 6;
  int n0 = (gw % ntiles) << 4;
  int lo = lane & 15, hi = lane >> 4;
  int n = n0 + lo;
  int y = n / WBD, x = n % WBD;

  v8f acc[4];
#pragma unroll
  for (int t = 0; t < 4; ++t)
#pragma unroll
    for (int r = 0; r < 8; ++r) acc[t][r] = bias[m0 + t * 16 + 8 * hi + r];

  const unsigned short* wbase = Wc + (size_t)(m0 + lo) * KCONV + 8 * hi;
#pragma unroll 4
  for (int k0 = 0; k0 < KCONV; k0 += 32) {
    Frag B;
    conv_chunk(img, y, x, k0 + 8 * hi,      &B.u[0]);
    conv_chunk(img, y, x, k0 + 16 + 8 * hi, &B.u[4]);
#pragma unroll
    for (int t = 0; t < 4; ++t) {
      Frag A;
      const unsigned short* wr = wbase + (size_t)t * 16 * KCONV + k0;
      *(uint4*)&A.u[0] = *(const uint4*)(wr);
      *(uint4*)&A.u[4] = *(const uint4*)(wr + 16);
      acc[t] = __builtin_amdgcn_wmma_f32_16x16x32_bf16(false, A.v, false, B.v,
                                                       (short)0, acc[t], false, false);
    }
  }
#pragma unroll
  for (int t = 0; t < 4; ++t) {
    int mbase = m0 + t * 16 + 8 * hi;
#pragma unroll
    for (int r = 0; r < 8; ++r) outF[(size_t)(mbase + r) * QN + n] = acc[t][r];
  }
}

// ---------------- (A + Badd) -> instance norm; writes f32 [c][q] and optional bf16 [q][c] ----------------
__global__ void add_instnorm_kernel(const float* __restrict__ A,
                                    const float* __restrict__ Badd,
                                    float* __restrict__ outF,
                                    unsigned short* __restrict__ outBF) {
  int c = blockIdx.x;
  const float* a = A + (size_t)c * QN;
  const float* b = Badd + (size_t)c * QN;
  float s = 0.f, ss = 0.f;
  for (int i = threadIdx.x; i < QN; i += blockDim.x) {
    float x = a[i] + b[i];
    s += x; ss += x * x;
  }
  __shared__ float rs[256], rss[256];
  rs[threadIdx.x] = s; rss[threadIdx.x] = ss;
  __syncthreads();
  for (int st = 128; st > 0; st >>= 1) {
    if ((int)threadIdx.x < st) { rs[threadIdx.x] += rs[threadIdx.x + st];
                                 rss[threadIdx.x] += rss[threadIdx.x + st]; }
    __syncthreads();
  }
  float mean = rs[0] * (1.f / QN);
  float var  = rss[0] * (1.f / QN) - mean * mean;
  float rstd = rsqrtf(var + 1e-5f);
  for (int i = threadIdx.x; i < QN; i += blockDim.x) {
    float y = (a[i] + b[i] - mean) * rstd;
    outF[(size_t)c * QN + i] = y;
    if (outBF) outBF[(size_t)i * DCH + c] = f2bf(y);
  }
}

// ---------------- offset/scale heads: sigmoid offsets, softmax scale-weights, points, refn, height ----------------
__global__ void heads_kernel(const unsigned short* __restrict__ qBF,
                             const unsigned short* __restrict__ tBF,
                             const float* __restrict__ bev_pos,
                             const float* __restrict__ off_w2, const float* __restrict__ off_b2,
                             const float* __restrict__ sw_w,   const float* __restrict__ sw_b,
                             float* __restrict__ points, float* __restrict__ refn,
                             float* __restrict__ swt, float* __restrict__ height_out) {
  int q = blockIdx.x * blockDim.x + threadIdx.x;
  if (q >= QN) return;
  const unsigned short* trow = tBF + (size_t)q * DCH;
  const unsigned short* qrow = qBF + (size_t)q * DCH;

  float offv[12];
#pragma unroll 1
  for (int j = 0; j < 12; ++j) {
    float acc = off_b2[j];
    const float* w = off_w2 + j * DCH;
    for (int k = 0; k < DCH; ++k) acc += w[k] * bf2f(trow[k]);
    offv[j] = 1.f / (1.f + expf(-acc));
  }
  float swl[16];
#pragma unroll 1
  for (int j = 0; j < 16; ++j) {
    float acc = sw_b[j];
    const float* w = sw_w + j * DCH;
    for (int k = 0; k < DCH; ++k) acc += w[k] * bf2f(qrow[k]);
    swl[j] = acc;
  }
#pragma unroll
  for (int p = 0; p < PPT; ++p) {
    float mx = swl[p*4];
    for (int l = 1; l < LLV; ++l) mx = fmaxf(mx, swl[p*4+l]);
    float e[4], sum = 0.f;
    for (int l = 0; l < LLV; ++l) { e[l] = expf(swl[p*4+l] - mx); sum += e[l]; }
    float inv = 1.f / sum;
    for (int l = 0; l < LLV; ++l) swt[(size_t)q*16 + p*4 + l] = e[l] * inv;
  }
  float ref[3];
#pragma unroll
  for (int d = 0; d < 3; ++d) {
    float s = 0.f;
    for (int i = 0; i < 4; ++i) s += bev_pos[((size_t)i * QN + q) * 3 + d];
    ref[d] = 0.25f * s;
  }
  const float lov[3] = {-50.f, -50.f, -4.f};
  const float rng[3] = {100.f, 100.f, 8.f};
#pragma unroll
  for (int p = 0; p < PPT; ++p) {
    float o0 = offv[p*3+0] * RXY * 2.f - RXY;
    float o1 = offv[p*3+1] * RXY * 2.f - RXY;
    float o2 = offv[p*3+2] * RZ  * 2.f - RZ;
    int row = q * PPT + p;
    float pw0 = ref[0] * rng[0] + lov[0] + o0;
    float pw1 = ref[1] * rng[1] + lov[1] + o1;
    float pw2 = ref[2] * rng[2] + lov[2] + o2;
    points[(size_t)row*3+0] = pw0;
    points[(size_t)row*3+1] = pw1;
    points[(size_t)row*3+2] = pw2;
    refn[(size_t)row*3+0] = (pw0 - lov[0]) / rng[0];
    refn[(size_t)row*3+1] = (pw1 - lov[1]) / rng[1];
    refn[(size_t)row*3+2] = (pw2 - lov[2]) / rng[2];
    height_out[(size_t)p * QN + q] = o2;
  }
}

// ---------------- pos MLP layer 1: relu(refn @ pos_w1^T + b1) -> bf16 [40000][256] ----------------
__global__ void pos1_kernel(const float* __restrict__ refn,
                            const float* __restrict__ w1, const float* __restrict__ b1,
                            unsigned short* __restrict__ posH) {
  int idx = blockIdx.x * blockDim.x + threadIdx.x;
  if (idx >= QN * PPT * 256) return;
  int row = idx >> 8, o = idx & 255;
  const float* r = refn + (size_t)row * 3;
  float acc = b1[o] + w1[o*3+0]*r[0] + w1[o*3+1]*r[1] + w1[o*3+2]*r[2];
  posH[(size_t)row * 256 + o] = f2bf(fmaxf(acc, 0.f));
}

// ---------------- sampling_4d: project to 6 cams, bilinear over 4 levels, level-softmax weighted ----------------
__global__ void sampling_kernel(const float* __restrict__ points,
                                const float* __restrict__ swt,
                                const float* __restrict__ l2i,
                                const float* __restrict__ f0, const float* __restrict__ f1,
                                const float* __restrict__ f2, const float* __restrict__ f3,
                                float* __restrict__ sampled) {
  int row = blockIdx.x;            // q*4 + p
  int c = threadIdx.x;             // channel
  int q = row >> 2, p = row & 3;
  float px = points[(size_t)row*3+0];
  float py = points[(size_t)row*3+1];
  float pz = points[(size_t)row*3+2];
  float wl[LLV];
#pragma unroll
  for (int l = 0; l < LLV; ++l) wl[l] = swt[(size_t)q*16 + p*4 + l];
  const float* feats[LLV] = {f0, f1, f2, f3};
  const int hhs[LLV] = {32, 16, 8, 4};
  const int wws[LLV] = {88, 44, 22, 11};

  float acc = 0.f;
  for (int n = 0; n < NCAM; ++n) {
    const float* M = l2i + n * 16;
    float cx = M[0]*px + M[1]*py + M[2]*pz  + M[3];
    float cy = M[4]*px + M[5]*py + M[6]*pz  + M[7];
    float cz = M[8]*px + M[9]*py + M[10]*pz + M[11];
    float zc = fmaxf(cz, 1e-5f);
    float u = cx / (zc * 704.f);
    float v = cy / (zc * 256.f);
    if (!(cz > 1e-5f && u > 0.f && u < 1.f && v > 0.f && v < 1.f)) continue;
#pragma unroll
    for (int l = 0; l < LLV; ++l) {
      int hh = hhs[l], ww = wws[l];
      float x = u * (float)ww - 0.5f;
      float y = v * (float)hh - 0.5f;
      float x0f = floorf(x), y0f = floorf(y);
      int x0 = (int)x0f, y0 = (int)y0f;
      float wx = x - x0f, wy = y - y0f;
      const float* F = feats[l] + (size_t)n * hh * ww * DCH;
      float wgt = wl[l];
      float w00 = (1.f - wx) * (1.f - wy) * wgt;
      float w10 = wx * (1.f - wy) * wgt;
      float w01 = (1.f - wx) * wy * wgt;
      float w11 = wx * wy * wgt;
      if (x0   >= 0 && x0   < ww && y0   >= 0 && y0   < hh) acc += F[((size_t)y0    *ww + x0  )*DCH + c] * w00;
      if (x0+1 >= 0 && x0+1 < ww && y0   >= 0 && y0   < hh) acc += F[((size_t)y0    *ww + x0+1)*DCH + c] * w10;
      if (x0   >= 0 && x0   < ww && y0+1 >= 0 && y0+1 < hh) acc += F[((size_t)(y0+1)*ww + x0  )*DCH + c] * w01;
      if (x0+1 >= 0 && x0+1 < ww && y0+1 >= 0 && y0+1 < hh) acc += F[((size_t)(y0+1)*ww + x0+1)*DCH + c] * w11;
    }
  }
  sampled[(size_t)row * DCH + c] = acc;
}

// ---------------- host-side orchestration ----------------
extern "C" void kernel_launch(void* const* d_in, const int* in_sizes, int n_in,
                              void* d_out, int out_size, void* d_ws, size_t ws_size,
                              hipStream_t stream) {
  (void)in_sizes; (void)n_in; (void)out_size; (void)ws_size;
  const float* feat0     = (const float*)d_in[0];
  const float* feat1     = (const float*)d_in[1];
  const float* feat2     = (const float*)d_in[2];
  const float* feat3     = (const float*)d_in[3];
  const float* lidar2img = (const float*)d_in[4];
  const float* bev_query = (const float*)d_in[5];
  const float* bev_pos   = (const float*)d_in[6];
  const float* in_w  = (const float*)d_in[7];   const float* in_b  = (const float*)d_in[8];
  const float* off_w1= (const float*)d_in[9];   const float* off_b1= (const float*)d_in[10];
  const float* off_w2= (const float*)d_in[11];  const float* off_b2= (const float*)d_in[12];
  const float* sw_w  = (const float*)d_in[13];  const float* sw_b  = (const float*)d_in[14];
  const float* pos_w1= (const float*)d_in[15];  const float* pos_b1= (const float*)d_in[16];
  const float* pos_w2= (const float*)d_in[17];  const float* pos_b2= (const float*)d_in[18];
  const float* mid_w1= (const float*)d_in[19];  const float* mid_b1= (const float*)d_in[20];
  const float* mid_w2= (const float*)d_in[21];  const float* mid_b2= (const float*)d_in[22];
  const float* mid_w3= (const float*)d_in[23];  const float* mid_b3= (const float*)d_in[24];
  const float* out_w = (const float*)d_in[25];  const float* out_b = (const float*)d_in[26];

  float* outQ = (float*)d_out;                       // (1,128,100,100)
  float* outH = (float*)d_out + DCH * QN;            // (1,1,4,100,100)

  // workspace carve-out
  char* base = (char*)d_ws;
  size_t off = 0;
  auto carve = [&](size_t bytes) -> void* {
    void* p = base + off;
    off += (bytes + 255) & ~(size_t)255;
    return p;
  };
  unsigned short* WinBF   = (unsigned short*)carve((size_t)DCH * KCONV * 2);
  unsigned short* WoutBF  = (unsigned short*)carve((size_t)DCH * KCONV * 2);
  unsigned short* offw1BF = (unsigned short*)carve((size_t)DCH * DCH * 2);
  unsigned short* midw1BF = (unsigned short*)carve((size_t)512 * 512 * 2);
  unsigned short* midw2BF = (unsigned short*)carve((size_t)512 * 512 * 2);
  unsigned short* midw3BF = (unsigned short*)carve((size_t)DCH * 512 * 2);
  unsigned short* posw2BF = (unsigned short*)carve((size_t)DCH * 256 * 2);
  unsigned short* qimgBF  = (unsigned short*)carve((size_t)QN * DCH * 2);
  float*          tmpF32  = (float*)carve((size_t)DCH * QN * 4);     // reused conv/GEMM f32 out
  float*          qF32    = (float*)carve((size_t)DCH * QN * 4);
  unsigned short* qBF     = (unsigned short*)carve((size_t)QN * DCH * 2);
  unsigned short* tBF     = (unsigned short*)carve((size_t)QN * DCH * 2);
  float*          pointsW = (float*)carve((size_t)QN * PPT * 3 * 4);
  float*          refnW   = (float*)carve((size_t)QN * PPT * 3 * 4);
  float*          swtW    = (float*)carve((size_t)QN * 16 * 4);
  unsigned short* posHBF  = (unsigned short*)carve((size_t)QN * PPT * 256 * 2);
  float*          sampled = (float*)carve((size_t)QN * PPT * DCH * 4);
  unsigned short* sfBF    = (unsigned short*)carve((size_t)QN * 512 * 2);  // == [40000][128] rows
  unsigned short* m1BF    = (unsigned short*)carve((size_t)QN * 512 * 2);
  unsigned short* m2BF    = (unsigned short*)carve((size_t)QN * 512 * 2);
  float*          q2F32   = (float*)carve((size_t)DCH * QN * 4);
  unsigned short* q2BF    = (unsigned short*)carve((size_t)QN * DCH * 2);

  const int T = 256;
  auto blks  = [](int n, int t) { return (n + t - 1) / t; };
  auto gblk  = [&](int waves) { return (waves * 32 + T - 1) / T; };  // grid for N waves

  // 1) weight conversions
  reorder_convw_kernel<<<blks(DCH*KCONV, T), T, 0, stream>>>(in_w,  WinBF);
  reorder_convw_kernel<<<blks(DCH*KCONV, T), T, 0, stream>>>(out_w, WoutBF);
  f32_to_bf16_kernel<<<blks(DCH*DCH, T), T, 0, stream>>>(off_w1, offw1BF, DCH*DCH);
  f32_to_bf16_kernel<<<blks(512*512, T), T, 0, stream>>>(mid_w1, midw1BF, 512*512);
  f32_to_bf16_kernel<<<blks(512*512, T), T, 0, stream>>>(mid_w2, midw2BF, 512*512);
  f32_to_bf16_kernel<<<blks(DCH*512, T), T, 0, stream>>>(mid_w3, midw3BF, DCH*512);
  f32_to_bf16_kernel<<<blks(DCH*256, T), T, 0, stream>>>(pos_w2, posw2BF, DCH*256);
  img_to_bf16_kernel<<<blks(DCH*QN, T), T, 0, stream>>>(bev_query, qimgBF);

  // 2) q = instance_norm(bev_query + conv3x3(bev_query))
  gemm_conv3x3_kernel<<<gblk((DCH/64)*(QN/16)), T, 0, stream>>>(WinBF, qimgBF, in_b, tmpF32);
  add_instnorm_kernel<<<DCH, T, 0, stream>>>(bev_query, tmpF32, qF32, qBF);

  // 3) t = relu(off_w1 @ q)
  gemm_bf16_kernel<<<gblk((DCH/64)*(QN/16)), T, 0, stream>>>(
      offw1BF, qBF, off_b1, nullptr, tBF, nullptr, DCH, QN, DCH, 1);

  // 4) offsets/scale-weights/points/refn/height
  heads_kernel<<<blks(QN, T), T, 0, stream>>>(qBF, tBF, bev_pos, off_w2, off_b2,
                                              sw_w, sw_b, pointsW, refnW, swtW, outH);

  // 5) pos layer 1, then sampling (independent)
  pos1_kernel<<<blks(QN*PPT*256, T), T, 0, stream>>>(refnW, pos_w1, pos_b1, posHBF);
  sampling_kernel<<<QN*PPT, DCH, 0, stream>>>(pointsW, swtW, lidar2img,
                                              feat0, feat1, feat2, feat3, sampled);

  // 6) sf = sampled + pos_w2 @ posH  (bf16, already in mid-GEMM layout [q][512])
  gemm_bf16_kernel<<<gblk((DCH/64)*((QN*PPT)/16)), T, 0, stream>>>(
      posw2BF, posHBF, pos_b2, sampled, sfBF, nullptr, DCH, QN*PPT, 256, 0);

  // 7) mid MLP (gelu, gelu, linear)
  gemm_bf16_kernel<<<gblk((512/64)*(QN/16)), T, 0, stream>>>(
      midw1BF, sfBF, mid_b1, nullptr, m1BF, nullptr, 512, QN, 512, 2);
  gemm_bf16_kernel<<<gblk((512/64)*(QN/16)), T, 0, stream>>>(
      midw2BF, m1BF, mid_b2, nullptr, m2BF, nullptr, 512, QN, 512, 2);
  gemm_bf16_kernel<<<gblk((DCH/64)*(QN/16)), T, 0, stream>>>(
      midw3BF, m2BF, mid_b3, nullptr, nullptr, tmpF32, DCH, QN, 512, 0);

  // 8) q = instance_norm(q + m); q = instance_norm(q + conv3x3(q))
  add_instnorm_kernel<<<DCH, T, 0, stream>>>(qF32, tmpF32, q2F32, q2BF);
  gemm_conv3x3_kernel<<<gblk((DCH/64)*(QN/16)), T, 0, stream>>>(WoutBF, q2BF, out_b, tmpF32);
  add_instnorm_kernel<<<DCH, T, 0, stream>>>(q2F32, tmpF32, outQ, nullptr);
}
